// CrossAtt_28561532519161
// MI455X (gfx1250) — matrix operand order repaired and checked by reading, not compiled, not run on hardware
//
#include <hip/hip_runtime.h>
#include <hip/hip_bf16.h>

// MI455X / gfx1250: wave32, WMMA 16x16x32 bf16 path.
typedef __attribute__((ext_vector_type(16))) __bf16 v16bf;
typedef __attribute__((ext_vector_type(8)))  float  v8f;

#define HD 256   // hidden size H
#define SL 512   // sequence length n = m
#define NB 2     // batch

// Native CDNA5 transcendental tanh (V_TANH_F32) when the builtin exists;
// fall back to libm so the build never breaks.
__device__ __forceinline__ float fast_tanh(float x) {
#if __has_builtin(__builtin_amdgcn_tanhf)
  return __builtin_amdgcn_tanhf(x);
#else
  return tanhf(x);
#endif
}

__device__ __forceinline__ float fast_rcp(float x) {
#if __has_builtin(__builtin_amdgcn_rcpf)
  return __builtin_amdgcn_rcpf(x);
#else
  return 1.0f / x;
#endif
}

// ---------------------------------------------------------------------------
// Pack a (256x256) f32 weight W into WMMA B-operand blocks (bf16), so the GEMM
// kernels load the B matrix with aligned 32B vector loads.
// B tile (K=32 x N=16) lane layout: n = lane%16, half = lane/16, pair v:
//   k = half*16 + 2*v  -> values W[n][k], W[n][k+1]   (B[k][n] = W[n][k], y=xW^T)
// pw layout: [nt(16)][kb(8)][lane(32)][pair(8)][2] bf16  (32768 u32 total)
// ---------------------------------------------------------------------------
__global__ __launch_bounds__(256) void pack_w_kernel(const float* __restrict__ W,
                                                     __bf16* __restrict__ pw) {
  int idx  = blockIdx.x * 256 + threadIdx.x;   // 0..32767
  int p    = idx & 7;
  int lane = (idx >> 3) & 31;
  int kb   = (idx >> 8) & 7;
  int nt   = idx >> 11;
  int n    = nt * 16 + (lane & 15);
  int half = lane >> 4;
  int k    = kb * 32 + half * 16 + 2 * p;
  pw[idx * 2 + 0] = (__bf16)W[n * HD + k + 0];
  pw[idx * 2 + 1] = (__bf16)W[n * HD + k + 1];
}

// ---------------------------------------------------------------------------
// C(1024x256) = A(1024x256, f32) @ W^T, W pre-packed as pw. One wave per
// 16x16 output tile: 8 bf16 WMMAs along K=256.
// A-operand lane layout: m = lane%16, half = lane/16, pair p:
//   k = half*16 + 2*p
// ---------------------------------------------------------------------------
__global__ __launch_bounds__(256) void gemm1024x256_kernel(const float* __restrict__ A,
    const __bf16* __restrict__ pw, float* __restrict__ C) {
  int wave = (blockIdx.x * 256 + threadIdx.x) >> 5;
  int lane = threadIdx.x & 31;
  int nt = wave & 15, mt = wave >> 4;          // mt in [0,64)
  int ln = lane & 15, half = lane >> 4;
  const float* arow = A + (mt * 16 + ln) * HD;
  v8f acc = {};
  for (int kb = 0; kb < 8; ++kb) {
    v16bf a;
#pragma unroll
    for (int p = 0; p < 8; ++p) {
      int k = kb * 32 + half * 16 + 2 * p;
      a[2 * p + 0] = (__bf16)arow[k + 0];
      a[2 * p + 1] = (__bf16)arow[k + 1];
    }
    v16bf bm = *(const v16bf*)(pw + ((size_t)(nt * 8 + kb) * 32 + lane) * 16);
    acc = __builtin_amdgcn_wmma_f32_16x16x32_bf16(false, a, false, bm,
                                                  (short)0, acc, false, false);
  }
#pragma unroll
  for (int r = 0; r < 8; ++r)
    C[(mt * 16 + r + 8 * half) * HD + nt * 16 + ln] = acc[r];
}

// ---------------------------------------------------------------------------
// Fused Bahdanau branch: for a tile of 16 (i,j) pairs (fixed b,i; 16 j's),
//   feat = tanh(q_proj[i,:] + k_proj[j,:])   (registers, native V_TANH_F32)
//   scores = feat @ Wv^T                      (16 N-tiles x 8 K-step WMMAs)
//   attn = softmax(scores, axis=H)            (register softmax, shfl within
//                                              the 16-lane C-matrix half)
// Never materializes feat/scores: only the 512MiB attn output hits HBM.
// One wave per tile; 32768 waves total.
// ---------------------------------------------------------------------------
__global__ __launch_bounds__(256) void bahdanau_kernel(const float* __restrict__ qp,
    const float* __restrict__ kp, const __bf16* __restrict__ pwv,
    float* __restrict__ attn) {
  int wave = (blockIdx.x * 256 + threadIdx.x) >> 5;
  int lane = threadIdx.x & 31;
  int jt = wave & 31;
  int i  = (wave >> 5) & (SL - 1);
  int b  = wave >> 14;
  int ln = lane & 15, half = lane >> 4;
  const float* qrow = qp + ((size_t)(b * SL) + i) * HD;
  const float* krow = kp + ((size_t)(b * SL) + jt * 16 + ln) * HD;

  v8f zero = {};
  v8f acc[16];
#pragma unroll
  for (int t = 0; t < 16; ++t) acc[t] = zero;

  for (int kb = 0; kb < 8; ++kb) {
    v16bf a;
#pragma unroll
    for (int p = 0; p < 8; ++p) {
      int k = kb * 32 + half * 16 + 2 * p;
      a[2 * p + 0] = (__bf16)fast_tanh(qrow[k + 0] + krow[k + 0]);
      a[2 * p + 1] = (__bf16)fast_tanh(qrow[k + 1] + krow[k + 1]);
    }
#pragma unroll
    for (int t = 0; t < 16; ++t) {
      v16bf bm = *(const v16bf*)(pwv + ((size_t)(t * 8 + kb) * 32 + lane) * 16);
      acc[t] = __builtin_amdgcn_wmma_f32_16x16x32_bf16(false, a, false, bm,
                                                       (short)0, acc[t], false, false);
    }
  }

  // Row-wise softmax over N=256 (16 tiles x 16 lanes within each half).
#pragma unroll
  for (int r = 0; r < 8; ++r) {
    float m = acc[0][r];
#pragma unroll
    for (int t = 1; t < 16; ++t) m = fmaxf(m, acc[t][r]);
#pragma unroll
    for (int x = 1; x < 16; x <<= 1) m = fmaxf(m, __shfl_xor(m, x, 32));
    float s = 0.f;
#pragma unroll
    for (int t = 0; t < 16; ++t) { float e = __expf(acc[t][r] - m); acc[t][r] = e; s += e; }
#pragma unroll
    for (int x = 1; x < 16; x <<= 1) s += __shfl_xor(s, x, 32);
    float inv = fast_rcp(s);
    size_t base = ((((size_t)b * SL) + i) * SL + jt * 16 + r + 8 * half) * HD + ln;
#pragma unroll
    for (int t = 0; t < 16; ++t) attn[base + t * 16] = acc[t][r] * inv;
  }
}

// ---------------------------------------------------------------------------
// Fused multi-head dot-product attention for one (batch*head, i-tile):
//   S = (Q/sqrt(d)) K^T  via WMMA -> 16x512 block staged in LDS,
//   softmax over j (full-wave reductions), P stored back to LDS,
//   O = P @ V via WMMA (A operand re-read from LDS in A layout).
// One wave per block (EXEC all ones), 32KB LDS, 512 blocks total.
// ---------------------------------------------------------------------------
__global__ __launch_bounds__(32) void mha_kernel(const float* __restrict__ qp,
    const float* __restrict__ kp, const float* __restrict__ vp,
    float* __restrict__ outh) {
  __shared__ float sb[16 * SL];
  int lane = threadIdx.x;
  int bh = blockIdx.x >> 5, it = blockIdx.x & 31;
  int b = bh >> 3, head = bh & 7;
  int ln = lane & 15, half = lane >> 4;
  const float scale = 0.17677669529663687f;  // 1/sqrt(32)

  // Q tile (16 x 32), pre-scaled, resident for all j tiles.
  v16bf aq;
  {
    const float* qrow = qp + ((size_t)(b * SL) + it * 16 + ln) * HD + head * 32;
#pragma unroll
    for (int p = 0; p < 8; ++p) {
      int k = half * 16 + 2 * p;
      aq[2 * p + 0] = (__bf16)(qrow[k + 0] * scale);
      aq[2 * p + 1] = (__bf16)(qrow[k + 1] * scale);
    }
  }

  // S = Q K^T, one WMMA per 16-wide j tile (K dim = d = 32).
  for (int jt = 0; jt < 32; ++jt) {
    const float* krow = kp + ((size_t)(b * SL) + jt * 16 + ln) * HD + head * 32;
    v16bf bk;
#pragma unroll
    for (int v = 0; v < 8; ++v) {
      int k = half * 16 + 2 * v;
      bk[2 * v + 0] = (__bf16)krow[k + 0];
      bk[2 * v + 1] = (__bf16)krow[k + 1];
    }
    v8f z = {};
    v8f s = __builtin_amdgcn_wmma_f32_16x16x32_bf16(false, aq, false, bk,
                                                    (short)0, z, false, false);
#pragma unroll
    for (int r = 0; r < 8; ++r)
      sb[(r + 8 * half) * SL + jt * 16 + ln] = s[r];
  }
  asm volatile("s_wait_dscnt 0x0" ::: "memory");

  // Softmax over j (512 cols per row) — whole wave cooperates per row.
  for (int r2 = 0; r2 < 16; ++r2) {
    float v[16];
#pragma unroll
    for (int u = 0; u < 16; ++u) v[u] = sb[r2 * SL + lane + 32 * u];
    float m = v[0];
#pragma unroll
    for (int u = 1; u < 16; ++u) m = fmaxf(m, v[u]);
#pragma unroll
    for (int x = 1; x < 32; x <<= 1) m = fmaxf(m, __shfl_xor(m, x, 32));
    float ssum = 0.f;
#pragma unroll
    for (int u = 0; u < 16; ++u) { v[u] = __expf(v[u] - m); ssum += v[u]; }
#pragma unroll
    for (int x = 1; x < 32; x <<= 1) ssum += __shfl_xor(ssum, x, 32);
    float inv = fast_rcp(ssum);
#pragma unroll
    for (int u = 0; u < 16; ++u) sb[r2 * SL + lane + 32 * u] = v[u] * inv;
  }
  asm volatile("s_wait_dscnt 0x0" ::: "memory");

  // O = P @ V : K=512 in 16 steps, N=32 as two 16-wide tiles.
  v8f co0 = {}, co1 = {};
  for (int kb = 0; kb < 16; ++kb) {
    v16bf ap;
#pragma unroll
    for (int p = 0; p < 8; ++p) {
      int k = kb * 32 + half * 16 + 2 * p;
      ap[2 * p + 0] = (__bf16)sb[ln * SL + k + 0];
      ap[2 * p + 1] = (__bf16)sb[ln * SL + k + 1];
    }
#pragma unroll
    for (int nh = 0; nh < 2; ++nh) {
      v16bf bv;
#pragma unroll
      for (int v = 0; v < 8; ++v) {
        int j = kb * 32 + half * 16 + 2 * v;
        const float* vr = vp + ((size_t)(b * SL) + j) * HD + head * 32 + nh * 16 + ln;
        bv[2 * v + 0] = (__bf16)vr[0];
        bv[2 * v + 1] = (__bf16)vr[HD];
      }
      if (nh == 0)
        co0 = __builtin_amdgcn_wmma_f32_16x16x32_bf16(false, ap, false, bv,
                                                      (short)0, co0, false, false);
      else
        co1 = __builtin_amdgcn_wmma_f32_16x16x32_bf16(false, ap, false, bv,
                                                      (short)0, co1, false, false);
    }
  }
#pragma unroll
  for (int r = 0; r < 8; ++r) {
    int row = (b * SL) + it * 16 + r + 8 * half;
    outh[(size_t)row * HD + head * 32 + 0  + ln] = co0[r];
    outh[(size_t)row * HD + head * 32 + 16 + ln] = co1[r];
  }
}

// ---------------------------------------------------------------------------
extern "C" void kernel_launch(void* const* d_in, const int* in_sizes, int n_in,
                              void* d_out, int out_size, void* d_ws, size_t ws_size,
                              hipStream_t stream) {
  (void)in_sizes; (void)n_in; (void)out_size; (void)ws_size;
  const float* queries = (const float*)d_in[0];
  const float* keys    = (const float*)d_in[1];
  // d_in[2] (values) is unused by the reference as well.
  const float* Wq = (const float*)d_in[3];
  const float* Wk = (const float*)d_in[4];
  const float* Wv = (const float*)d_in[5];
  const float* Wo = (const float*)d_in[6];

  // Workspace layout (4.5 MB): packed weights (4 x 128KB bf16), then f32
  // q_proj / k_proj / v_proj / head-output (1 MB each).
  __bf16* pWq = (__bf16*)d_ws;
  __bf16* pWk = pWq + 65536;
  __bf16* pWv = pWk + 65536;
  __bf16* pWo = pWv + 65536;
  float* qproj = (float*)(pWo + 65536);
  float* kproj = qproj + NB * SL * HD;
  float* vproj = kproj + NB * SL * HD;
  float* outh  = vproj + NB * SL * HD;

  float* out0 = (float*)d_out;          // (2,512,256) final output
  float* attn = out0 + NB * SL * HD;    // (2,512,512,256) attn_weights

  pack_w_kernel<<<128, 256, 0, stream>>>(Wq, pWq);
  pack_w_kernel<<<128, 256, 0, stream>>>(Wk, pWk);
  pack_w_kernel<<<128, 256, 0, stream>>>(Wv, pWv);
  pack_w_kernel<<<128, 256, 0, stream>>>(Wo, pWo);

  gemm1024x256_kernel<<<128, 256, 0, stream>>>(queries, pWq, qproj);
  gemm1024x256_kernel<<<128, 256, 0, stream>>>(keys,    pWk, kproj);
  gemm1024x256_kernel<<<128, 256, 0, stream>>>(keys,    pWv, vproj);

  bahdanau_kernel<<<4096, 256, 0, stream>>>(qproj, kproj, pWv, attn);
  mha_kernel<<<512, 32, 0, stream>>>(qproj, kproj, vproj, outh);
  gemm1024x256_kernel<<<128, 256, 0, stream>>>(outh, pWo, out0);
}
